// Multi_DGCN_24232205484262
// MI455X (gfx1250) — matrix operand order repaired and checked by reading, not compiled, not run on hardware
//
#include <hip/hip_runtime.h>

typedef __attribute__((ext_vector_type(16))) __bf16 v16bf;
typedef __attribute__((ext_vector_type(8)))  float  v8f;

#define B_  32
#define N_  512
#define D_  768
#define L_  2
#define NBR_ 4
#define FH_ 256
#define NC_ 2
#define GAMMA_ (-0.1f)
#define ZETA_  (1.1f)

// ---------------------------------------------------------------------------
// WMMA fragment loaders (bf16, 16x16x32).  ISA layouts (CDNA5 ISA):
//   A 16x32: lane m=l&15; e<8 -> K=e, e>=8 -> K=e+8; lanes 16-31 add K+=8
//   B 32x16: lane n=l&15; e<8 -> K=e, e>=8 -> K=e+8; lanes 16-31 add K+=8
//   C 16x16 f32: lane n=l&15; vgpr j -> M=j (+8 for lanes 16-31)
// A is M-major (stride lda), B is N-major (stride ldb) so K-pairs are
// contiguous -> compiler merges the 16-bit element loads into b128 loads.
// ---------------------------------------------------------------------------
__device__ __forceinline__ v16bf load_a_frag(const __bf16* __restrict__ A,
                                             int mBase, int kBase, int lda, int Mmax) {
  int lane = threadIdx.x & 31;
  int m = mBase + (lane & 15);
  if (m >= Mmax) m = Mmax - 1;                 // clamp: keeps reads in-bounds
  const __bf16* row = A + (size_t)m * lda + kBase + ((lane & 16) ? 8 : 0);
  v16bf a;
#pragma unroll
  for (int e = 0; e < 16; ++e) {
    int k = (e < 8) ? e : (e + 8);
    a[e] = row[k];
  }
  return a;
}

__device__ __forceinline__ v16bf load_b_frag(const __bf16* __restrict__ Bt,
                                             int nBase, int kBase, int ldb) {
  int lane = threadIdx.x & 31;
  int n = nBase + (lane & 15);
  const __bf16* col = Bt + (size_t)n * ldb + kBase + ((lane & 16) ? 8 : 0);
  v16bf b;
#pragma unroll
  for (int e = 0; e < 16; ++e) b[e] = col[(e < 8) ? e : (e + 8)];
  return b;
}

// ---------------------------------------------------------------------------
// Generic batched WMMA GEMM:  C[M,N] = scale * A[M,K] * B[K,N]
// A: bf16 M-major (lda), Bt: bf16 N-major (ldb), C: f32 or bf16 (ldc).
// Block = 128 threads (4 waves, 2x2), block tile 128x128, wave tile 64x64
// (4x4 accumulator tiles -> 0.5 KB cache traffic per WMMA).
// ---------------------------------------------------------------------------
template <bool OUT_BF16>
__global__ void __launch_bounds__(128)
wmma_gemm_kernel(const __bf16* __restrict__ Aall, const __bf16* __restrict__ Btall,
                 void* __restrict__ Call, int M, int N, int K,
                 long long aBatch, long long bBatch, long long cBatch,
                 int lda, int ldb, int ldc, float scale) {
  int bz = blockIdx.z;
  const __bf16* A  = Aall  + (size_t)bz * aBatch;
  const __bf16* Bt = Btall + (size_t)bz * bBatch;

  int wave = threadIdx.x >> 5;
  int wr = wave >> 1, wc = wave & 1;
  int mBase = blockIdx.y * 128 + wr * 64;
  int nBase = blockIdx.x * 128 + wc * 64;

  v8f acc[4][4] = {};

  for (int k = 0; k < K; k += 32) {
    if (k + 32 < K) {  // gfx1250 global_prefetch_b8 hints for the next K tile
      int lane = threadIdx.x & 31;
      int mp = mBase + (lane & 15); if (mp >= M) mp = M - 1;
      __builtin_prefetch(A  + (size_t)mp * lda + k + 32, 0, 3);
      __builtin_prefetch(Bt + (size_t)(nBase + (lane & 15)) * ldb + k + 32, 0, 3);
    }
    v16bf af[4], bfr[4];
#pragma unroll
    for (int i = 0; i < 4; ++i) af[i]  = load_a_frag(A,  mBase + 16 * i, k, lda, M);
#pragma unroll
    for (int j = 0; j < 4; ++j) bfr[j] = load_b_frag(Bt, nBase + 16 * j, k, ldb);
#pragma unroll
    for (int i = 0; i < 4; ++i)
#pragma unroll
      for (int j = 0; j < 4; ++j)
        acc[i][j] = __builtin_amdgcn_wmma_f32_16x16x32_bf16(
            false, af[i], false, bfr[j], (short)0, acc[i][j], false, false);
  }

  int lane = threadIdx.x & 31;
  int nl = lane & 15;
  int mh = (lane & 16) ? 8 : 0;
#pragma unroll
  for (int ti = 0; ti < 4; ++ti)
#pragma unroll
    for (int tj = 0; tj < 4; ++tj) {
      int col = nBase + tj * 16 + nl;
#pragma unroll
      for (int j = 0; j < 8; ++j) {
        int row = mBase + ti * 16 + j + mh;
        if (row < M && col < N) {
          float v = acc[ti][tj][j] * scale;
          if constexpr (OUT_BF16)
            ((__bf16*)Call)[(size_t)bz * cBatch + (size_t)row * ldc + col] = (__bf16)v;
          else
            ((float*)Call)[(size_t)bz * cBatch + (size_t)row * ldc + col] = v;
        }
      }
    }
}

// ---------------------------------------------------------------------------
// Elementwise f32 -> bf16
// ---------------------------------------------------------------------------
__global__ void f32_to_bf16_kernel(const float* __restrict__ in, __bf16* __restrict__ out, size_t n) {
  size_t i = (size_t)blockIdx.x * blockDim.x + threadIdx.x;
  if (i < n) out[i] = (__bf16)in[i];
}

// ---------------------------------------------------------------------------
// LDS-tiled transpose (batched): in (b, R, C) TIN  ->  out (b, C, R) bf16
// ---------------------------------------------------------------------------
template <typename TIN>
__global__ void transpose_to_bf16_kernel(const TIN* __restrict__ in, __bf16* __restrict__ out,
                                         int R, int C, long long inBatch, long long outBatch) {
  __shared__ float tile[32][33];
  int b = blockIdx.z;
  int c0 = blockIdx.x * 32, r0 = blockIdx.y * 32;
  int tx = threadIdx.x, ty = threadIdx.y;
  for (int i = ty; i < 32; i += 8) {
    int r = r0 + i, c = c0 + tx;
    tile[i][tx] = (r < R && c < C) ? (float)in[(size_t)b * inBatch + (size_t)r * C + c] : 0.f;
  }
  __syncthreads();
  for (int i = ty; i < 32; i += 8) {
    int c = c0 + i, r = r0 + tx;
    if (c < C && r < R)
      out[(size_t)b * outBatch + (size_t)c * R + r] = (__bf16)tile[tx][i];
  }
}

// ---------------------------------------------------------------------------
// Gate + degree: in-place logits -> adj (fp32), dis[b,i] = deg>0 ? deg^-0.5 : 0
// One block per (b, i) row.
// ---------------------------------------------------------------------------
__global__ void __launch_bounds__(256)
gate_deg_kernel(float* __restrict__ logits, const float* __restrict__ pmask,
                float* __restrict__ dis) {
  int row = blockIdx.x;              // row = b*N_ + qi
  int b = row / N_, qi = row - b * N_;
  float mq = pmask[(size_t)b * N_ + qi];
  float* lrow = logits + (size_t)row * N_;
  const float* mrow = pmask + (size_t)b * N_;
  float sum = 0.f;
  for (int j = threadIdx.x; j < N_; j += 256) {
    float lg = lrow[j];
    float s  = 1.f / (1.f + __expf(-lg));
    float g  = s * (ZETA_ - GAMMA_) + GAMMA_;
    g = fminf(fmaxf(g, 0.f), 1.f);
    float adj = g * mq * mrow[j];
    lrow[j] = adj;
    sum += adj;
  }
  __shared__ float red[256];
  red[threadIdx.x] = sum;
  __syncthreads();
  for (int s = 128; s > 0; s >>= 1) {
    if (threadIdx.x < s) red[threadIdx.x] += red[threadIdx.x + s];
    __syncthreads();
  }
  if (threadIdx.x == 0) {
    float deg = red[0];
    dis[row] = (deg > 0.f) ? rsqrtf(deg) : 0.f;
  }
}

// ---------------------------------------------------------------------------
// Normalize + transpose:  adjT_bf16[b][j][i] = dis[i] * adj[b][i][j] * dis[j]
// ---------------------------------------------------------------------------
__global__ void norm_transpose_kernel(const float* __restrict__ adj, const float* __restrict__ dis,
                                      __bf16* __restrict__ adjT) {
  __shared__ float tile[32][33];
  int b = blockIdx.z;
  int j0 = blockIdx.x * 32, i0 = blockIdx.y * 32;
  const float* adjb = adj + (size_t)b * N_ * N_;
  const float* disb = dis + (size_t)b * N_;
  int tx = threadIdx.x, ty = threadIdx.y;
  for (int t = ty; t < 32; t += 8) {
    int i = i0 + t, j = j0 + tx;
    tile[t][tx] = disb[i] * adjb[(size_t)i * N_ + j] * disb[j];
  }
  __syncthreads();
  __bf16* outb = adjT + (size_t)b * N_ * N_;
  for (int t = ty; t < 32; t += 8) {
    int j = j0 + t, i = i0 + tx;
    outb[(size_t)j * N_ + i] = (__bf16)tile[tx][t];
  }
}

// ---------------------------------------------------------------------------
// Row extraction (node N-1 of each layer's features)
// ---------------------------------------------------------------------------
__global__ void extract_row_f32_kernel(const float* __restrict__ feat, float* __restrict__ outp,
                                       int rowIdx, int rows, int Dd) {
  int b = blockIdx.x;
  const float* src = feat + ((size_t)b * rows + rowIdx) * Dd;
  for (int d = threadIdx.x; d < Dd; d += blockDim.x) outp[(size_t)b * Dd + d] = src[d];
}

__global__ void extract_row_bf16_kernel(const __bf16* __restrict__ feat, float* __restrict__ outp,
                                        int rowIdx, int rows, int Dd) {
  int b = blockIdx.x;
  const __bf16* src = feat + ((size_t)b * rows + rowIdx) * Dd;
  for (int d = threadIdx.x; d < Dd; d += blockDim.x) outp[(size_t)b * Dd + d] = (float)src[d];
}

// ---------------------------------------------------------------------------
// Per-branch head: retain-gated sum -> fc1(relu) -> fc2(relu) -> fcf
// One block per batch element.
// ---------------------------------------------------------------------------
__global__ void __launch_bounds__(256)
head_kernel(const float* __restrict__ p0, const float* __restrict__ p1, const float* __restrict__ p2,
            const float* __restrict__ pw, const float* __restrict__ pb,
            const float* __restrict__ w1, const float* __restrict__ b1,
            const float* __restrict__ w2, const float* __restrict__ b2,
            const float* __restrict__ wf, const float* __restrict__ bf,
            float* __restrict__ out, int branch) {
  int b = blockIdx.x, t = threadIdx.x;
  __shared__ float x[D_];
  __shared__ float h1[FH_];
  __shared__ float h2[FH_];
  __shared__ float red[256];
  __shared__ float retain[3];
  const float* ps[3] = { p0 + (size_t)b * D_, p1 + (size_t)b * D_, p2 + (size_t)b * D_ };

  for (int l = 0; l < 3; ++l) {
    float s = 0.f;
    for (int d = t; d < D_; d += 256) s += ps[l][d] * pw[d];
    red[t] = s;
    __syncthreads();
    for (int st = 128; st > 0; st >>= 1) {
      if (t < st) red[t] += red[t + st];
      __syncthreads();
    }
    if (t == 0) retain[l] = 1.f / (1.f + __expf(-(red[0] + pb[0])));
    __syncthreads();
  }
  for (int d = t; d < D_; d += 256)
    x[d] = retain[0] * ps[0][d] + retain[1] * ps[1][d] + retain[2] * ps[2][d];
  __syncthreads();
  {
    float s = b1[t];
    const float* w = w1 + (size_t)t * D_;
    for (int d = 0; d < D_; ++d) s += x[d] * w[d];
    h1[t] = fmaxf(s, 0.f);
  }
  __syncthreads();
  {
    float s = b2[t];
    const float* w = w2 + (size_t)t * FH_;
    for (int d = 0; d < FH_; ++d) s += h1[d] * w[d];
    h2[t] = fmaxf(s, 0.f);
  }
  __syncthreads();
  if (t < NC_) {
    float s = bf[t];
    const float* w = wf + (size_t)t * FH_;
    for (int g = 0; g < FH_; ++g) s += h2[g] * w[g];
    out[(size_t)b * NBR_ * NC_ + (size_t)branch * NC_ + t] = s;
  }
}

// ---------------------------------------------------------------------------
// Host-side orchestration
// ---------------------------------------------------------------------------
static inline void launch_gemm(bool outBf16, const __bf16* A, const __bf16* Bt, void* C,
                               int M, int N, int K,
                               long long aB, long long bB, long long cB,
                               int lda, int ldb, int ldc, float scale, int batch,
                               hipStream_t s) {
  dim3 grid((N + 127) / 128, (M + 127) / 128, batch), block(128);
  if (outBf16)
    wmma_gemm_kernel<true><<<grid, block, 0, s>>>(A, Bt, C, M, N, K, aB, bB, cB, lda, ldb, ldc, scale);
  else
    wmma_gemm_kernel<false><<<grid, block, 0, s>>>(A, Bt, C, M, N, K, aB, bB, cB, lda, ldb, ldc, scale);
}

extern "C" void kernel_launch(void* const* d_in, const int* in_sizes, int n_in,
                              void* d_out, int out_size, void* d_ws, size_t ws_size,
                              hipStream_t stream) {
  (void)in_sizes; (void)n_in; (void)out_size; (void)ws_size;
  const float* pmask   = (const float*)d_in[0];
  const float* feature = (const float*)d_in[1];
  const float* Wq      = (const float*)d_in[2];
  const float* Wk      = (const float*)d_in[3];
  const float* proj_w  = (const float*)d_in[4];
  const float* proj_b  = (const float*)d_in[5];
  const float* fc1_w   = (const float*)d_in[6];
  const float* fc1_b   = (const float*)d_in[7];
  const float* fc2_w   = (const float*)d_in[8];
  const float* fc2_b   = (const float*)d_in[9];
  const float* fcf_w   = (const float*)d_in[10];
  const float* fcf_b   = (const float*)d_in[11];
  float* out = (float*)d_out;

  char* ws = (char*)d_ws;
  size_t off = 0;
  auto alloc = [&](size_t bytes) -> void* {
    void* p = ws + off;
    off = (off + bytes + 255) & ~(size_t)255;
    return p;
  };

  const size_t FEAT_E = (size_t)B_ * N_ * D_;   // 12.6M elems
  const size_t ADJ_E  = (size_t)B_ * N_ * N_;   // 8.4M elems
  const size_t W_E    = (size_t)NBR_ * L_ * D_ * D_;

  __bf16* WqT   = (__bf16*)alloc(W_E * 2);
  __bf16* WkT   = (__bf16*)alloc(W_E * 2);
  __bf16* F0    = (__bf16*)alloc(FEAT_E * 2);
  __bf16* F0T   = (__bf16*)alloc(FEAT_E * 2);
  __bf16* F1    = (__bf16*)alloc(FEAT_E * 2);
  __bf16* F1T   = (__bf16*)alloc(FEAT_E * 2);
  __bf16* Qb    = (__bf16*)alloc(FEAT_E * 2);
  __bf16* Kb    = (__bf16*)alloc(FEAT_E * 2);
  float*  LOG   = (float*)alloc(ADJ_E * 4);      // logits -> adj (in place)
  __bf16* ADJT  = (__bf16*)alloc(ADJ_E * 2);
  float*  DIS   = (float*)alloc((size_t)B_ * N_ * 4);
  __bf16* FBAND = (__bf16*)alloc((size_t)B_ * 16 * D_ * 2);
  float*  P0    = (float*)alloc((size_t)B_ * D_ * 4);
  float*  P1    = (float*)alloc((size_t)B_ * D_ * 4);
  float*  P2    = (float*)alloc((size_t)B_ * D_ * 4);

  dim3 tb(32, 8);
  const float invSqrtD = 1.0f / sqrtf((float)D_);

  // One-time conversions
  transpose_to_bf16_kernel<float><<<dim3(D_ / 32, D_ / 32, NBR_ * L_), tb, 0, stream>>>(
      Wq, WqT, D_, D_, (long long)D_ * D_, (long long)D_ * D_);
  transpose_to_bf16_kernel<float><<<dim3(D_ / 32, D_ / 32, NBR_ * L_), tb, 0, stream>>>(
      Wk, WkT, D_, D_, (long long)D_ * D_, (long long)D_ * D_);
  f32_to_bf16_kernel<<<(unsigned)((FEAT_E + 255) / 256), 256, 0, stream>>>(feature, F0, FEAT_E);
  transpose_to_bf16_kernel<float><<<dim3(D_ / 32, N_ / 32, B_), tb, 0, stream>>>(
      feature, F0T, N_, D_, (long long)N_ * D_, (long long)D_ * N_);
  extract_row_f32_kernel<<<B_, 256, 0, stream>>>(feature, P0, N_ - 1, N_, D_);

  for (int br = 0; br < NBR_; ++br) {
    for (int l = 0; l < L_; ++l) {
      const __bf16* feat  = (l == 0) ? F0  : F1;
      const __bf16* featT = (l == 0) ? F0T : F1T;
      const __bf16* wq = WqT + ((size_t)br * L_ + l) * D_ * D_;
      const __bf16* wk = WkT + ((size_t)br * L_ + l) * D_ * D_;

      // q = feat * Wq  (flat M = B*N, weight shared across batch)
      launch_gemm(true, feat, wq, Qb, B_ * N_, D_, D_, 0, 0, 0, D_, D_, D_, 1.f, 1, stream);
      // k = feat * Wk
      launch_gemm(true, feat, wk, Kb, B_ * N_, D_, D_, 0, 0, 0, D_, D_, D_, 1.f, 1, stream);
      // logits[b] = (q[b] . k[b]^T) / sqrt(D)   (k row-major (n,d) == N-major B)
      launch_gemm(false, Qb, Kb, LOG, N_, N_, D_,
                  (long long)N_ * D_, (long long)N_ * D_, (long long)N_ * N_,
                  D_, D_, N_, invSqrtD, B_, stream);
      // gate + degree (in place), then normalized transposed adjacency in bf16
      gate_deg_kernel<<<B_ * N_, 256, 0, stream>>>(LOG, pmask, DIS);
      norm_transpose_kernel<<<dim3(N_ / 32, N_ / 32, B_), tb, 0, stream>>>(LOG, DIS, ADJT);

      if (l == 0) {
        // F1[b] = adjT[b] * feat[b]   (B = featT, N-major)
        launch_gemm(true, ADJT, featT, F1, N_, D_, N_,
                    (long long)N_ * N_, (long long)D_ * N_, (long long)N_ * D_,
                    N_, N_, D_, 1.f, B_, stream);
        transpose_to_bf16_kernel<__bf16><<<dim3(D_ / 32, N_ / 32, B_), tb, 0, stream>>>(
            F1, F1T, N_, D_, (long long)N_ * D_, (long long)D_ * N_);
        extract_row_bf16_kernel<<<B_, 256, 0, stream>>>(F1, P1, N_ - 1, N_, D_);
      } else {
        // Head only reads node N-1 -> compute just the 16-row band [496, 512)
        launch_gemm(true, ADJT + 496 * N_, featT, FBAND, 16, D_, N_,
                    (long long)N_ * N_, (long long)D_ * N_, (long long)16 * D_,
                    N_, N_, D_, 1.f, B_, stream);
        extract_row_bf16_kernel<<<B_, 256, 0, stream>>>(FBAND, P2, 15, 16, D_);
      }
    }

    head_kernel<<<B_, 256, 0, stream>>>(
        P0, P1, P2,
        proj_w + (size_t)br * D_, proj_b + br,
        fc1_w + (size_t)br * FH_ * D_, fc1_b + (size_t)br * FH_,
        fc2_w + (size_t)br * FH_ * FH_, fc2_b + (size_t)br * FH_,
        fcf_w + (size_t)br * NC_ * FH_, fcf_b + (size_t)br * NC_,
        out, br);
  }
}
